// NodeGLAM_738734375079
// MI455X (gfx1250) — matrix operand order repaired and checked by reading, not compiled, not run on hardware
//
#include <hip/hip_runtime.h>

// ---------------------------------------------------------------------------
// NodeGLAM for MI455X (gfx1250, wave32, WMMA + TDM).
// bf16 activations + v_wmma_f32_16x16x32_bf16 GEMMs, fp32 L2 atomics for the
// graph scatter (accumulator fits in 192MB L2), TDM-staged A tiles in LDS.
// ---------------------------------------------------------------------------

typedef __attribute__((ext_vector_type(16))) __bf16 v16bf;
typedef __attribute__((ext_vector_type(8)))  __bf16 v8bf;
typedef __attribute__((ext_vector_type(8)))  float  v8f;
typedef __attribute__((ext_vector_type(4)))  unsigned int v4u;
typedef __attribute__((ext_vector_type(8)))  int v8i;
typedef __attribute__((ext_vector_type(4)))  int v4i;

#define CDIV(a, b) (((a) + (b) - 1) / (b))

// TDM (tensor_load_to_lds) availability: device pass only, guarded.
#if defined(__HIP_DEVICE_COMPILE__) && defined(__has_builtin)
# if __has_builtin(__builtin_amdgcn_tensor_load_to_lds) && \
     __has_builtin(__builtin_amdgcn_s_wait_tensorcnt)
#  define USE_TDM 1
# endif
#endif

__device__ __forceinline__ unsigned short f2bf(float f) {
  union { float f; unsigned u; } v; v.f = f;
  unsigned r = v.u + 0x7FFFu + ((v.u >> 16) & 1u);   // round-to-nearest-even
  return (unsigned short)(r >> 16);
}
__device__ __forceinline__ float bf2f(unsigned short u) {
  union { unsigned u; float f; } v; v.u = ((unsigned)u) << 16;
  return v.f;
}
__device__ __forceinline__ float gelu_f(float x) {
  return 0.5f * x * (1.0f + erff(x * 0.70710678118654752f));
}

// ---------------------------- small utility kernels ------------------------

__global__ void zero_kernel(float* __restrict__ p, size_t n) {
  size_t i = (size_t)blockIdx.x * blockDim.x + threadIdx.x;
  size_t stride = (size_t)gridDim.x * blockDim.x;
  for (; i < n; i += stride) p[i] = 0.0f;
}

__global__ void fill_bias_kernel(float* __restrict__ C, const float* __restrict__ b,
                                 size_t n, int NC) {
  size_t i = (size_t)blockIdx.x * blockDim.x + threadIdx.x;
  if (i < n) C[i] = b[i % NC];
}

__global__ void tobf_kernel(const float* __restrict__ in, unsigned short* __restrict__ out,
                            size_t n) {
  size_t i = (size_t)blockIdx.x * blockDim.x + threadIdx.x;
  if (i < n) out[i] = f2bf(in[i]);
}

__global__ void gelu2bf_kernel(const float* __restrict__ in, unsigned short* __restrict__ out,
                               size_t n) {
  size_t i = (size_t)blockIdx.x * blockDim.x + threadIdx.x;
  if (i < n) out[i] = f2bf(gelu_f(in[i]));
}

// ---------------------------- graph preprocessing --------------------------

__global__ void degree_kernel(const int* __restrict__ col, int E, float* __restrict__ deg) {
  int e = blockIdx.x * blockDim.x + threadIdx.x;
  if (e < E) atomicAdd(&deg[col[e]], 1.0f);
}

__device__ __forceinline__ float deg_inv_sqrt(float d) {
  return d > 0.0f ? rsqrtf(fmaxf(d, 1.0f)) : 0.0f;
}

__global__ void edge_norm_kernel(const int* __restrict__ row, const int* __restrict__ col,
                                 const float* __restrict__ deg, int E,
                                 float* __restrict__ norm) {
  int e = blockIdx.x * blockDim.x + threadIdx.x;
  if (e < E) norm[e] = deg_inv_sqrt(deg[row[e]]) * deg_inv_sqrt(deg[col[e]]);
}

// ---------------------------- batch norm -----------------------------------

__global__ void bn_partial_kernel(const float* __restrict__ x, int N,
                                  float* __restrict__ sum, float* __restrict__ sumsq) {
  int c = threadIdx.x;                       // 128 threads, one per channel
  int chunk = CDIV(N, (int)gridDim.x);
  int r0 = blockIdx.x * chunk;
  int r1 = r0 + chunk; if (r1 > N) r1 = N;
  float s = 0.0f, q = 0.0f;
  for (int r = r0; r < r1; ++r) {
    float v = x[(size_t)r * 128 + c];
    s += v; q += v * v;
  }
  atomicAdd(&sum[c], s);
  atomicAdd(&sumsq[c], q);
}

__global__ void bn_apply_kernel(const float* __restrict__ x, const float* __restrict__ sum,
                                const float* __restrict__ sumsq,
                                const float* __restrict__ gamma, const float* __restrict__ beta,
                                int N, unsigned short* __restrict__ xn) {
  size_t i = (size_t)blockIdx.x * blockDim.x + threadIdx.x;
  if (i >= (size_t)N * 128) return;
  int c = (int)(i & 127);
  float invN = 1.0f / (float)N;
  float mu = sum[c] * invN;
  float var = sumsq[c] * invN - mu * mu;
  float sc = gamma[c] * rsqrtf(var + 1e-5f);
  float sh = beta[c] - mu * sc;
  xn[i] = f2bf(x[i] * sc + sh);
}

// ---------------------------- weight repack (fp32 -> WMMA B-fragment bf16) --
// Packed layout (dword index t):  t = ((kt*nct + ct)*32 + lane)*8 + v
//   lane 0-15  hold K = kt*32 + {0..15}  (pairs 2v,2v+1)
//   lane 16-31 hold K = kt*32 + {16..31}
//   column = ct*16 + (lane & 15)
__global__ void repack_kernel(const float* __restrict__ W, unsigned* __restrict__ out, int NC) {
  int t = blockIdx.x * blockDim.x + threadIdx.x;
  int total = 64 * NC;                        // KD=128 -> 128*NC/2 dwords
  if (t >= total) return;
  int v = t & 7;
  int lane = (t >> 3) & 31;
  int rest = t >> 8;
  int nct = NC >> 4;
  int ct = rest % nct;
  int kt = rest / nct;
  int k = kt * 32 + ((lane & 16) ? 16 : 0) + 2 * v;
  int c = ct * 16 + (lane & 15);
  unsigned lo = f2bf(W[(size_t)k * NC + c]);
  unsigned hi = f2bf(W[(size_t)(k + 1) * NC + c]);
  out[t] = lo | (hi << 16);
}

// ---------------------------- WMMA GEMM ------------------------------------
// C[N,NC] += A[N,128] @ W[128,NC].  A bf16 row-major, W in packed-B layout.
// One block = 16 rows; one wave = one 16x16 output tile (blockDim = 32*NC/16).
// A tile (16 rows x 128 bf16 = 4KB contiguous) staged into LDS via the
// Tensor Data Mover (TDM) when available, else a synchronous vector copy.
__global__ void gemm_wmma_kernel(const unsigned short* __restrict__ A,
                                 const unsigned short* __restrict__ Wp,
                                 float* __restrict__ C, int NC) {
  __shared__ __align__(16) unsigned short As[16 * 128];   // 4KB A tile
  const int tid = threadIdx.x;
  const int rowBase = blockIdx.x * 16;

#if defined(USE_TDM)
  if (tid < 32) {                      // wave 0 issues one TDM descriptor
    unsigned long long ga = (unsigned long long)(uintptr_t)(A + (size_t)rowBase * 128);
    unsigned ldsa = (unsigned)(uintptr_t)(&As[0]);
    // D# group 0: count=1 | lds_addr | global_addr[56:0] | type=2
    v4u g0;
    g0[0] = 1u;                                            // count=1, user mode
    g0[1] = ldsa;                                          // lds_addr
    g0[2] = (unsigned)ga;                                  // global_addr[31:0]
    g0[3] = (unsigned)((ga >> 32) & 0x01FFFFFFu) | (2u << 30);  // [56:32] | type=2
    // D# group 1: data_size=2B; tensor 128x16 (dim0=128, dim1=16);
    // tile 128x16; tensor_dim0_stride=128  -> contiguous 4KB transfer
    v8i g1 = { (int)0x00010000,   // data_size=1 (2 bytes) at [17:16]
               (int)0x00800000,   // tensor_dim0 lo16 (=128) at [63:48]
               (int)0x00100000,   // dim0 hi16=0, tensor_dim1 lo16 (=16) at [95:80]
               (int)0x00800000,   // dim1 hi16=0, tile_dim0 (=128) at [127:112]
               16,                // tile_dim1=16, tile_dim2=0
               128,               // tensor_dim0_stride lo32 = 128
               0, 0 };            // stride hi, tensor_dim1_stride = 0
    v4i z4 = {};
#if __clang_major__ >= 23
    v8i z8 = {};
    __builtin_amdgcn_tensor_load_to_lds(g0, g1, z4, z4, z8, 0);
#else
    __builtin_amdgcn_tensor_load_to_lds(g0, g1, z4, z4, 0);
#endif
    __builtin_amdgcn_s_wait_tensorcnt(0);
  }
  __syncthreads();
#else
  {
    const uint4* src = (const uint4*)(A + (size_t)rowBase * 128);
    uint4* dst = (uint4*)As;
    for (int i = tid; i < 256; i += blockDim.x) dst[i] = src[i];
  }
  __syncthreads();
#endif

  const int lane = tid & 31;
  const int ct   = tid >> 5;                 // column tile == wave id
  const int m    = lane & 15;                // A row within tile / C column idx
  const int half = (lane >> 4) << 3;         // 0 for lanes 0-15, 8 for 16-31
  const int nct  = NC >> 4;

  // C fragment (fp32 read-modify-write accumulate), 32-bit offsets
  const unsigned cbase = (unsigned)((rowBase + half) * NC + ct * 16 + m);
  v8f acc;
#pragma unroll
  for (int r = 0; r < 8; ++r) acc[r] = C[cbase + (unsigned)(r * NC)];

#pragma unroll
  for (int kt = 0; kt < 4; ++kt) {
    // A fragment: lane<16 -> K {0..15}, lane>=16 -> K {16..31}
    const unsigned short* ap = &As[m * 128 + kt * 32 + 2 * half];
    v8bf lo = *(const v8bf*)ap;
    v8bf hi = *(const v8bf*)(ap + 8);
    v16bf a = __builtin_shufflevector(lo, hi, 0, 1, 2, 3, 4, 5, 6, 7,
                                      8, 9, 10, 11, 12, 13, 14, 15);
    // B fragment: contiguous 32B per lane from packed weights
    v16bf b = *(const v16bf*)(Wp + (((unsigned)(kt * nct + ct) * 32u + lane) << 4));
    acc = __builtin_amdgcn_wmma_f32_16x16x32_bf16(false, a, false, b,
                                                  (short)0, acc, false, false);
  }

#pragma unroll
  for (int r = 0; r < 8; ++r) C[cbase + (unsigned)(r * NC)] = acc[r];
}

// ---------------------------- graph propagation ----------------------------
// out[col[e], :] += norm[e] * h[row[e], :]   (fp32 atomics, accumulator in L2)
__global__ void prop_kernel(const int* __restrict__ row, const int* __restrict__ col,
                            const float* __restrict__ norm,
                            const unsigned short* __restrict__ hin,
                            float* __restrict__ out, int E) {
  const int tx = threadIdx.x;                           // channel 0..127
  const int e = blockIdx.x * blockDim.y + threadIdx.y;  // edge
  if (e >= E) return;
  if (e + 8 < E)                                        // global_prefetch_b8
    __builtin_prefetch(&hin[(unsigned)row[e + 8] * 128u + tx], 0, 1);
  float nv = norm[e];
  if (nv != 0.0f) {
    unsigned r = (unsigned)row[e], c = (unsigned)col[e];
    float v = nv * bf2f(hin[r * 128u + tx]);
    atomicAdd(&out[c * 128u + tx], v);
  }
}

// ---------------------------- heads ----------------------------------------

__global__ void softmax64_kernel(const float* __restrict__ apre,
                                 float* __restrict__ out, int N) {
  const int wave = threadIdx.x >> 5;
  const int lane = threadIdx.x & 31;
  const int rowi = blockIdx.x * (blockDim.x >> 5) + wave;
  if (rowi >= N) return;
  const float* p = apre + (size_t)rowi * 64;
  float v0 = p[lane], v1 = p[lane + 32];
  float m = fmaxf(v0, v1);
  for (int off = 16; off > 0; off >>= 1) m = fmaxf(m, __shfl_xor(m, off, 32));
  float e0 = expf(v0 - m), e1 = expf(v1 - m);
  float s = e0 + e1;
  for (int off = 16; off > 0; off >>= 1) s += __shfl_xor(s, off, 32);
  float inv = 1.0f / s;
  out[(size_t)rowi * 64 + lane] = e0 * inv;
  out[(size_t)rowi * 64 + lane + 32] = e1 * inv;
}

__global__ void classifier_kernel(const float* __restrict__ apre,
                                  const float* __restrict__ wc,
                                  const float* __restrict__ bc,
                                  float* __restrict__ out, int N) {
  __shared__ float wcs[64 * 8];
  __shared__ float bcs[8];
  for (int i = threadIdx.x; i < 512; i += blockDim.x) wcs[i] = wc[i];
  if (threadIdx.x < 8) bcs[threadIdx.x] = bc[threadIdx.x];
  __syncthreads();
  int r = blockIdx.x * blockDim.x + threadIdx.x;
  if (r >= N) return;
  const float* p = apre + (size_t)r * 64;
  float acc[8];
#pragma unroll
  for (int j = 0; j < 8; ++j) acc[j] = bcs[j];
  for (int k = 0; k < 64; ++k) {
    float g = gelu_f(p[k]);
#pragma unroll
    for (int j = 0; j < 8; ++j) acc[j] += g * wcs[k * 8 + j];
  }
#pragma unroll
  for (int j = 0; j < 8; ++j) out[(size_t)r * 8 + j] = acc[j];
}

// ---------------------------------------------------------------------------
// Host orchestration
// ---------------------------------------------------------------------------

extern "C" void kernel_launch(void* const* d_in, const int* in_sizes, int n_in,
                              void* d_out, int out_size, void* d_ws, size_t ws_size,
                              hipStream_t stream) {
  const float* x     = (const float*)d_in[0];
  const int*   ei    = (const int*)d_in[1];
  const float* gamma = (const float*)d_in[2];
  const float* beta  = (const float*)d_in[3];
  const float* w1    = (const float*)d_in[4];
  const float* b1    = (const float*)d_in[5];
  const float* tag1w = (const float*)d_in[6];
  const float* tag1b = (const float*)d_in[7];
  const float* w2    = (const float*)d_in[8];
  const float* b2    = (const float*)d_in[9];
  const float* tag2w = (const float*)d_in[10];
  const float* tag2b = (const float*)d_in[11];
  const float* w5    = (const float*)d_in[12];
  const float* b5    = (const float*)d_in[13];
  const float* w6    = (const float*)d_in[14];
  const float* b6    = (const float*)d_in[15];
  const float* wc    = (const float*)d_in[16];
  const float* bc    = (const float*)d_in[17];

  const int N = in_sizes[0] / 128;
  const int E = in_sizes[1] / 2;
  const int* erow = ei;
  const int* ecol = ei + E;
  const size_t ND = (size_t)N * 128;

  // ---- workspace carve-up (~187 MB total) ----
  char* ws = (char*)d_ws;
  size_t off = 0;
  auto carve = [&](size_t bytes) -> void* {
    void* p = ws + off;
    off = (off + bytes + 255) & ~(size_t)255;
    return p;
  };
  float* deg    = (float*)carve((size_t)N * 4);
  float* nrm    = (float*)carve((size_t)E * 4);
  float* colsum = (float*)carve(256 * 4);           // [0:128) sum, [128:256) sumsq
  float* colsq  = colsum + 128;
  unsigned short* w1p   = (unsigned short*)carve(16384 * 2);
  unsigned short* tag1p = (unsigned short*)carve(4 * 16384 * 2);
  unsigned short* w2p   = (unsigned short*)carve(16384 * 2);
  unsigned short* tag2p = (unsigned short*)carve(4 * 16384 * 2);
  unsigned short* w5ap  = (unsigned short*)carve(16384 * 2);
  unsigned short* w5bp  = (unsigned short*)carve(16384 * 2);
  unsigned short* w6p   = (unsigned short*)carve(8192 * 2);
  unsigned short* xn_bf = (unsigned short*)carve(ND * 2);
  unsigned short* hA    = (unsigned short*)carve(ND * 2);
  unsigned short* hP    = (unsigned short*)carve(ND * 2);
  float* accC = (float*)carve(ND * 4);
  float* accP = (float*)carve(ND * 4);
  float* apre = accP;                                // reused after last hop

  float* out_a  = (float*)d_out;                     // [N,64] softmax
  float* out_cl = out_a + (size_t)N * 64;            // [N,8] classifier

  // ---- graph norm ----
  zero_kernel<<<1024, 256, 0, stream>>>(deg, (size_t)N);
  degree_kernel<<<CDIV(E, 256), 256, 0, stream>>>(ecol, E, deg);
  edge_norm_kernel<<<CDIV(E, 256), 256, 0, stream>>>(erow, ecol, deg, E, nrm);

  // ---- batch norm ----
  zero_kernel<<<1, 256, 0, stream>>>(colsum, 256);
  bn_partial_kernel<<<240, 128, 0, stream>>>(x, N, colsum, colsq);
  bn_apply_kernel<<<CDIV(ND, 256), 256, 0, stream>>>(x, colsum, colsq, gamma, beta, N, xn_bf);

  // ---- repack weights into WMMA B layout ----
  repack_kernel<<<32, 256, 0, stream>>>(w1, (unsigned*)w1p, 128);
  for (int k = 0; k < 4; ++k)
    repack_kernel<<<32, 256, 0, stream>>>(tag1w + (size_t)k * 16384, (unsigned*)(tag1p + (size_t)k * 16384), 128);
  repack_kernel<<<32, 256, 0, stream>>>(w2, (unsigned*)w2p, 128);
  for (int k = 0; k < 4; ++k)
    repack_kernel<<<32, 256, 0, stream>>>(tag2w + (size_t)k * 16384, (unsigned*)(tag2p + (size_t)k * 16384), 128);
  repack_kernel<<<32, 256, 0, stream>>>(w5, (unsigned*)w5ap, 128);            // rows 0..127
  repack_kernel<<<32, 256, 0, stream>>>(w5 + 16384, (unsigned*)w5bp, 128);    // rows 128..255
  repack_kernel<<<16, 256, 0, stream>>>(w6, (unsigned*)w6p, 64);

  const int rowTiles = N / 16;
  auto gemm128 = [&](const unsigned short* Ain, const unsigned short* Wp, float* Cout) {
    gemm_wmma_kernel<<<rowTiles, 256, 0, stream>>>(Ain, Wp, Cout, 128);
  };
  auto tag_conv = [&](const unsigned short* h0, const unsigned short* wp, const float* bias) {
    fill_bias_kernel<<<CDIV(ND, 256), 256, 0, stream>>>(accC, bias, ND, 128);
    gemm128(h0, wp, accC);
    const unsigned short* src = h0;
    for (int k = 1; k <= 3; ++k) {
      zero_kernel<<<2048, 256, 0, stream>>>(accP, ND);
      prop_kernel<<<CDIV(E, 4), dim3(128, 4), 0, stream>>>(erow, ecol, nrm, src, accP, E);
      tobf_kernel<<<CDIV(ND, 256), 256, 0, stream>>>(accP, hP, ND);
      gemm128(hP, wp + (size_t)k * 16384, accC);
      src = hP;
    }
  };

  // ---- layer 1: gelu(xn @ w1 + b1) ----
  fill_bias_kernel<<<CDIV(ND, 256), 256, 0, stream>>>(accC, b1, ND, 128);
  gemm128(xn_bf, w1p, accC);
  gelu2bf_kernel<<<CDIV(ND, 256), 256, 0, stream>>>(accC, hA, ND);

  // ---- TAGConv 1 + gelu ----
  tag_conv(hA, tag1p, tag1b);
  gelu2bf_kernel<<<CDIV(ND, 256), 256, 0, stream>>>(accC, hA, ND);

  // ---- layer 2: gelu(h @ w2 + b2) ----
  fill_bias_kernel<<<CDIV(ND, 256), 256, 0, stream>>>(accC, b2, ND, 128);
  gemm128(hA, w2p, accC);
  gelu2bf_kernel<<<CDIV(ND, 256), 256, 0, stream>>>(accC, hA, ND);

  // ---- TAGConv 2 + gelu ----
  tag_conv(hA, tag2p, tag2b);
  gelu2bf_kernel<<<CDIV(ND, 256), 256, 0, stream>>>(accC, hA, ND);

  // ---- concat [xn, h] @ w5 + b5, gelu ----
  fill_bias_kernel<<<CDIV(ND, 256), 256, 0, stream>>>(accC, b5, ND, 128);
  gemm128(xn_bf, w5ap, accC);
  gemm128(hA, w5bp, accC);
  gelu2bf_kernel<<<CDIV(ND, 256), 256, 0, stream>>>(accC, hP, ND);

  // ---- a_pre = h @ w6 + b6  (NC = 64) ----
  const size_t N64 = (size_t)N * 64;
  fill_bias_kernel<<<CDIV(N64, 256), 256, 0, stream>>>(apre, b6, N64, 64);
  gemm_wmma_kernel<<<rowTiles, 128, 0, stream>>>(hP, w6p, apre, 64);

  // ---- heads ----
  softmax64_kernel<<<CDIV(N, 8), 256, 0, stream>>>(apre, out_a, N);
  classifier_kernel<<<CDIV(N, 256), 256, 0, stream>>>(apre, wc, bc, out_cl, N);
}
